// LoRALinear_79955111182814
// MI455X (gfx1250) — compile-verified, matrix operand order checked by reading, not run on hardware
//
#include <hip/hip_runtime.h>
#include <hip/hip_bf16.h>

typedef __bf16 bf16;
typedef bf16  v16bf  __attribute__((ext_vector_type(16)));
typedef bf16  bf16x8 __attribute__((ext_vector_type(8)));
typedef float v8f    __attribute__((ext_vector_type(8)));
typedef float f32x4  __attribute__((ext_vector_type(4)));
typedef unsigned u32x4 __attribute__((ext_vector_type(4)));
typedef int      i32x4 __attribute__((ext_vector_type(4)));
typedef int      i32x8 __attribute__((ext_vector_type(8)));

#define D_IN   4096
#define D_OUT  4096
#define M_TOT  8192
#define LORA_K 16
#define LORA_S 2.0f   // 32.0 / 16

__device__ __forceinline__ void split_bf16(float f, bf16& h, bf16& l) {
    unsigned u = __float_as_uint(f);
    h = __builtin_bit_cast(bf16, (unsigned short)(u >> 16));   // truncated hi
    float hf = __uint_as_float(u & 0xffff0000u);
    l = (bf16)(f - hf);                                        // RNE residual
}

// ---------------------------------------------------------------------------
// Kernel 1: fold LoRA into the weight, emit bf16 hi/lo split.
// ---------------------------------------------------------------------------
__global__ __launch_bounds__(256) void lora_fold(
    const float* __restrict__ W, const float* __restrict__ L,
    const float* __restrict__ R, bf16* __restrict__ wh, bf16* __restrict__ wl)
{
    size_t idx = (size_t)blockIdx.x * 256 + threadIdx.x;   // 16M elements
    int o = (int)(idx >> 12);
    int i = (int)(idx & (D_IN - 1));
    float acc = W[idx];
#pragma unroll
    for (int j = 0; j < LORA_K; ++j)
        acc = fmaf(LORA_S * L[o * LORA_K + j], R[j * D_IN + i], acc);
    bf16 h, l;
    split_bf16(acc, h, l);
    wh[idx] = h;
    wl[idx] = l;
}

// ---------------------------------------------------------------------------
// TDM: async-load a 128 x 32 bf16 tile (row stride 4096 elem) into LDS,
// padding 4 dwords after every 16 dwords -> LDS row stride 40 bf16 (80B).
// D# group layout per CDNA5 ISA ch. 8 (CDNA5 ISA).
// ---------------------------------------------------------------------------
#define LDT 40   // LDS row stride in bf16

__device__ __forceinline__ void tdm_load_tile(unsigned lds_addr, const bf16* gptr) {
    unsigned long long ga = (unsigned long long)(uintptr_t)gptr;
    u32x4 g0;
    g0[0] = 1u;                                            // count=1, user desc
    g0[1] = lds_addr;                                      // LDS byte address
    g0[2] = (unsigned)(ga & 0xffffffffu);                  // global_addr lo
    g0[3] = (unsigned)((ga >> 32) & 0x01ffffffu) | (2u << 30);  // ga hi | type=2
    i32x8 g1;
    g1[0] = (int)((1u << 16)    // data_size = 2 bytes
                | (1u << 20)    // pad_enable
                | (3u << 22)    // pad_interval: 16 dwords (64B = one tile row)
                | (3u << 25));  // pad_amount: 4 dwords (16B)
    g1[1] = (int)(4096u << 16);  // tensor_dim0 = 4096 (bits 79:48)
    g1[2] = (int)(4096u << 16);  // tensor_dim1 = 4096 (bits 111:80)
    g1[3] = (int)(32u << 16);    // tile_dim0 = 32 elements (bits 127:112)
    g1[4] = 128;                 // tile_dim1 = 128 rows, tile_dim2 = 0
    g1[5] = 4096;                // tensor_dim0_stride = 4096 elements
    g1[6] = 0;
    g1[7] = 0;
    i32x4 z4 = {};
#if __clang_major__ >= 23
    i32x8 z8 = {};
    __builtin_amdgcn_tensor_load_to_lds(g0, g1, z4, z4, z8, 0);
#else
    __builtin_amdgcn_tensor_load_to_lds(g0, g1, z4, z4, 0);
#endif
}

// ---------------------------------------------------------------------------
// Kernel 2: bf16x3 WMMA GEMM  Y = X * W_eff^T + bias
// 128x128 block, BK=32, double-buffered LDS; B tiles via TDM, A tiles via
// VGPR path (fp32 -> bf16 hi/lo split). 24 v_wmma per k-step.
// ---------------------------------------------------------------------------
__global__ __launch_bounds__(256) void lora_gemm(
    const float* __restrict__ X, const bf16* __restrict__ Wh,
    const bf16* __restrict__ Wl, const float* __restrict__ bias,
    float* __restrict__ Y)
{
    __shared__ alignas(16) bf16 sAh[2][128][LDT];
    __shared__ alignas(16) bf16 sAl[2][128][LDT];
    __shared__ alignas(16) bf16 sBh[2][128][LDT];
    __shared__ alignas(16) bf16 sBl[2][128][LDT];

    const int tid  = threadIdx.x;
    const int lane = tid & 31;
    const int wave = tid >> 5;
    const int wm   = wave & 3;          // 0..3 : 32-row slab of M
    const int wn   = wave >> 2;         // 0..1 : 64-col slab of N
    const int lr   = lane & 15;         // row within 16x16 tile
    const int ks   = (lane >> 4) * 8;   // K phase (0 or 8), ISA 16-bit layout
    const int hi8  = (lane >> 4) * 8;   // C/D: lanes 16..31 hold M+8

    const int m0 = blockIdx.y * 128;
    const int n0 = blockIdx.x * 128;

    // A-loader mapping: 2 threads per row, 16 fp32 each
    const int arow = tid >> 1;
    const int acol = (tid & 1) * 16;
    const float* gA = X + (size_t)(m0 + arow) * D_IN + acol;

    // B tile bases (row 0 of the 128-row N slab)
    const bf16* gBh0 = Wh + (size_t)n0 * D_IN;
    const bf16* gBl0 = Wl + (size_t)n0 * D_IN;

    v8f c[2][4];
    {
        v8f z = {};
#pragma unroll
        for (int mi = 0; mi < 2; ++mi)
#pragma unroll
            for (int ni = 0; ni < 4; ++ni) c[mi][ni] = z;
    }

    // per-lane fragment base pointers into buffer 0
    const bf16* pAh = &sAh[0][wm * 32 + lr][ks];
    const bf16* pAl = &sAl[0][wm * 32 + lr][ks];
    const bf16* pBh = &sBh[0][wn * 64 + lr][ks];
    const bf16* pBl = &sBl[0][wn * 64 + lr][ks];
    const int BUFE = 128 * LDT;          // elements per buffer

    union FragU { v16bf v; bf16x8 h[2]; };

    // ---- prologue: fill buffer 0 ----
    if (wave == 0) {
        tdm_load_tile((unsigned)(uintptr_t)&sBh[0][0][0], gBh0);
        tdm_load_tile((unsigned)(uintptr_t)&sBl[0][0][0], gBl0);
    }
    {
        union { f32x4 v[4]; float e[16]; } a;
#pragma unroll
        for (int q = 0; q < 4; ++q) a.v[q] = *(const f32x4*)(gA + q * 4);
        union { bf16x8 v[2]; bf16 e[16]; } h, l;
#pragma unroll
        for (int i = 0; i < 16; ++i) split_bf16(a.e[i], h.e[i], l.e[i]);
        *(bf16x8*)&sAh[0][arow][acol]     = h.v[0];
        *(bf16x8*)&sAh[0][arow][acol + 8] = h.v[1];
        *(bf16x8*)&sAl[0][arow][acol]     = l.v[0];
        *(bf16x8*)&sAl[0][arow][acol + 8] = l.v[1];
    }
    if (wave == 0) __builtin_amdgcn_s_wait_tensorcnt(0);
    __syncthreads();

    // ---- main loop: stage k+32 into nxt while computing from cur ----
#pragma unroll 1
    for (int k0 = 0; k0 < D_IN; k0 += 32) {
        const int cur = (k0 >> 5) & 1;
        const int nxt = cur ^ 1;
        const int kk  = k0 + 32;

        if (kk < D_IN) {
            if (wave == 0) {
                tdm_load_tile((unsigned)(uintptr_t)&sBh[nxt][0][0], gBh0 + kk);
                tdm_load_tile((unsigned)(uintptr_t)&sBl[nxt][0][0], gBl0 + kk);
            }
            union { f32x4 v[4]; float e[16]; } a;
#pragma unroll
            for (int q = 0; q < 4; ++q)
                a.v[q] = *(const f32x4*)(gA + kk + q * 4);
            union { bf16x8 v[2]; bf16 e[16]; } h, l;
#pragma unroll
            for (int i = 0; i < 16; ++i) split_bf16(a.e[i], h.e[i], l.e[i]);
            *(bf16x8*)&sAh[nxt][arow][acol]     = h.v[0];
            *(bf16x8*)&sAh[nxt][arow][acol + 8] = h.v[1];
            *(bf16x8*)&sAl[nxt][arow][acol]     = l.v[0];
            *(bf16x8*)&sAl[nxt][arow][acol + 8] = l.v[1];
            if (kk + 32 < D_IN) __builtin_prefetch(gA + kk + 32, 0, 0);
        }

        // ---- fragments from LDS buffer `cur` ----
        const int bo = cur * BUFE;
        FragU ah[2], al[2], bh[4], bl[4];
#pragma unroll
        for (int mi = 0; mi < 2; ++mi) {
            const bf16* p = pAh + bo + mi * 16 * LDT;
            ah[mi].h[0] = *(const bf16x8*)p;
            ah[mi].h[1] = *(const bf16x8*)(p + 16);
            const bf16* q = pAl + bo + mi * 16 * LDT;
            al[mi].h[0] = *(const bf16x8*)q;
            al[mi].h[1] = *(const bf16x8*)(q + 16);
        }
#pragma unroll
        for (int ni = 0; ni < 4; ++ni) {
            const bf16* p = pBh + bo + ni * 16 * LDT;
            bh[ni].h[0] = *(const bf16x8*)p;
            bh[ni].h[1] = *(const bf16x8*)(p + 16);
            const bf16* q = pBl + bo + ni * 16 * LDT;
            bl[ni].h[0] = *(const bf16x8*)q;
            bl[ni].h[1] = *(const bf16x8*)(q + 16);
        }

        // ---- bf16x3: hh + hl + lh (24 wmma) ----
#pragma unroll
        for (int mi = 0; mi < 2; ++mi)
#pragma unroll
            for (int ni = 0; ni < 4; ++ni) {
                c[mi][ni] = __builtin_amdgcn_wmma_f32_16x16x32_bf16(
                    false, ah[mi].v, false, bh[ni].v, (short)0, c[mi][ni], false, false);
                c[mi][ni] = __builtin_amdgcn_wmma_f32_16x16x32_bf16(
                    false, ah[mi].v, false, bl[ni].v, (short)0, c[mi][ni], false, false);
                c[mi][ni] = __builtin_amdgcn_wmma_f32_16x16x32_bf16(
                    false, al[mi].v, false, bh[ni].v, (short)0, c[mi][ni], false, false);
            }

        // TDM completion (wave 0 owns TENSORcnt), then publish to all waves
        if (wave == 0) __builtin_amdgcn_s_wait_tensorcnt(0);
        __syncthreads();
    }

    // ---- epilogue: bias add + store (C/D layout: VGPR r -> M = r + hi8) ----
#pragma unroll
    for (int ni = 0; ni < 4; ++ni) {
        int n = n0 + wn * 64 + ni * 16 + lr;
        float bv = bias[n];
#pragma unroll
        for (int mi = 0; mi < 2; ++mi) {
            int mbase = m0 + wm * 32 + mi * 16 + hi8;
#pragma unroll
            for (int r = 0; r < 8; ++r)
                Y[(size_t)(mbase + r) * D_OUT + n] = c[mi][ni][r] + bv;
        }
    }
}

// ---------------------------------------------------------------------------
extern "C" void kernel_launch(void* const* d_in, const int* in_sizes, int n_in,
                              void* d_out, int out_size, void* d_ws, size_t ws_size,
                              hipStream_t stream) {
    const float* x    = (const float*)d_in[0];   // [4,2048,4096]
    const float* W    = (const float*)d_in[1];   // [4096,4096]
    const float* bias = (const float*)d_in[2];   // [4096]
    const float* L    = (const float*)d_in[3];   // [4096,16]
    const float* R    = (const float*)d_in[4];   // [16,4096]
    float* out = (float*)d_out;                  // [8192,4096]

    bf16* wh = (bf16*)d_ws;                      // 32 MB
    bf16* wl = wh + (size_t)D_OUT * D_IN;        // 32 MB

    lora_fold<<<(D_OUT * (size_t)D_IN) / 256, 256, 0, stream>>>(W, L, R, wh, wl);

    dim3 grid(D_OUT / 128, M_TOT / 128);
    lora_gemm<<<grid, 256, 0, stream>>>(x, wh, wl, bias, out);
}
